// MEANDock_24283745091642
// MI455X (gfx1250) — matrix operand order repaired and checked by reading, not compiled, not run on hardware
//
#include <hip/hip_runtime.h>
#include <hip/hip_bf16.h>
#include <cstdint>
#include <cstddef>

// ---------------------------------------------------------------------------
// MEANDock forward for MI455X (gfx1250, wave32, WMMA).
// Per-edge MLP GEMMs run on v_wmma_f32_16x16x32_bf16 with f32 accumulation.
// A tiles are staged into LDS via global_load_async_to_lds_b128 (ASYNCcnt,
// double buffered) and shared by all 8 n-tile waves of a block; weights are
// pre-swizzled into the exact B-operand VGPR image (pure b128 loads + wmma).
//
// Assumed d_in order (dict insertion order, params = jax tree_leaves):
//  0:X  1:S  2:generate_mask(u8)  3:position_ids  4:segment_ids  5:lengths
//  6:atom_mask(u8)  7:aa_emb
//  8+22*l+{0:att_ctx 1:att_inter 2..5:coord_ctx(W0,b0,W1,b1) 6..9:coord_inter
//          10..13:edge_ctx 14..17:edge_inter 18..21:node}   for l=0..2
//  74:pos_emb  75:round_emb
// ---------------------------------------------------------------------------

typedef __bf16 bf16;
typedef __attribute__((ext_vector_type(16))) __bf16 v16bf;
typedef __attribute__((ext_vector_type(8)))  float  v8f;

union V16 { v16bf v; uint4 q[2]; };

constexpr int B_   = 32, L_ = 256, C_ = 4, HID = 128, K_ = 9;
constexpr int N_   = B_ * L_;        // 8192
constexpr int NK   = N_ * K_;        // 73728
constexpr int EIN  = 2 * HID + C_ * C_;   // 272
constexpr int EINP = 288;            // padded to multiple of 32
constexpr int NLAYERS = 3, NITER = 3;
constexpr int LSTRIDE = 40;          // LDS A-tile row stride in bf16 (80 B)

static __device__ __forceinline__ bf16 tobf(float x) {
  union { float f; unsigned u; } c; c.f = x;
  unsigned r = c.u + 0x7fffu + ((c.u >> 16) & 1u);   // RNE
  unsigned short h = (unsigned short)(r >> 16);
  return __builtin_bit_cast(bf16, h);
}
static __device__ __forceinline__ float siluf(float x) {
  return x / (1.f + __expf(-x));
}

// ---------------------------------------------------------------------------
// Weight pre-swizzle: W[K,N] f32 -> Bpack[nt][kt][lane][16] bf16, where
// lane's 16 contiguous bf16 are B[kt*32 + 16*(lane>>4) + kk, nt*16 + (lane&15)]
// ---------------------------------------------------------------------------
__global__ void pack_b_kernel(const float* __restrict__ W, bf16* __restrict__ out,
                              int Kreal, int Kp, int Nn) {
  int idx = blockIdx.x * blockDim.x + threadIdx.x;
  int ktn = Kp >> 5;
  int total = (Nn >> 4) * ktn * 512;
  if (idx >= total) return;
  int kk   = idx & 15;
  int lane = (idx >> 4) & 31;
  int tile = idx >> 9;
  int kt   = tile % ktn;
  int nt   = tile / ktn;
  int k = (kt << 5) + ((lane >> 4) << 4) + kk;
  int n = (nt << 4) + (lane & 15);
  float w = (k < Kreal) ? W[(size_t)k * Nn + n] : 0.f;
  out[idx] = tobf(w);
}

// ---------------------------------------------------------------------------
// WMMA GEMM with async-LDS staged A: block = 8 waves = one 64-row m-block,
// wave w -> n-tile (blockIdx.y*8 + w).  Double-buffered A tile in LDS.
// act: 0 = identity, 1 = SiLU.  outF / outB optional (f32 / bf16).
// ---------------------------------------------------------------------------
__global__ void gemm_bf16_wmma(const bf16* __restrict__ A, const bf16* __restrict__ Bp,
                               const float* __restrict__ bias,
                               float* __restrict__ outF, bf16* __restrict__ outB,
                               int M, int Kp, int Nn, int act) {
  __shared__ bf16 tile[2][64 * LSTRIDE];        // 2 x 64 x 32 (+pad) = 10 KB
  int tid  = threadIdx.x;                       // 256 threads
  int wv   = tid >> 5, lane = tid & 31;
  int nT   = Nn >> 4;
  int mBase = blockIdx.x << 6;
  int nt   = blockIdx.y * 8 + wv;
  bool active = (nt < nT);
  int half = lane >> 4, l16 = lane & 15;
  int ktn  = Kp >> 5;

  // each thread async-copies one b128 chunk of the 64x32 A tile per k-step
  int srow = tid >> 2, schunk = tid & 3;
  const bf16* gsrc = A + (size_t)(mBase + srow) * Kp + (schunk << 3);
  unsigned lbase0 = (unsigned)(size_t)(&tile[0][srow * LSTRIDE]) + (schunk << 4);
  unsigned lbase1 = (unsigned)(size_t)(&tile[1][srow * LSTRIDE]) + (schunk << 4);

  auto stage = [&](int kt, int buf) {
    const bf16* g = gsrc + (kt << 5);
    unsigned laddr = buf ? lbase1 : lbase0;
    asm volatile("global_load_async_to_lds_b128 %0, %1, off"
                 :: "v"(laddr), "v"(g) : "memory");
  };

  v8f acc[4] = {};
  const bf16* bpL = Bp + ((size_t)nt * ktn) * 512 + (size_t)lane * 16;

  stage(0, 0);
  asm volatile("s_wait_asynccnt 0x0" ::: "memory");
  __syncthreads();

  for (int kt = 0; kt < ktn; ++kt) {
    int buf = kt & 1;
    if (kt + 1 < ktn) stage(kt + 1, buf ^ 1);

    if (active) {
      V16 b;
      const uint4* pb = (const uint4*)(bpL + (size_t)kt * 512);
      b.q[0] = pb[0]; b.q[1] = pb[1];
      if (kt + 1 < ktn) __builtin_prefetch(bpL + (size_t)(kt + 1) * 512, 0, 1);
#pragma unroll
      for (int mi = 0; mi < 4; ++mi) {
        const bf16* lrow = &tile[buf][((mi << 4) + l16) * LSTRIDE + (half << 4)];
        V16 a;
        a.q[0] = *(const uint4*)lrow;          // K +0..7
        a.q[1] = *(const uint4*)(lrow + 8);    // K +8..15
        acc[mi] = __builtin_amdgcn_wmma_f32_16x16x32_bf16(
            false, a.v, false, b.v, (short)0, acc[mi], false, false);
      }
    }
    // producers of buf^1 done + consumers of buf done before next overwrite
    asm volatile("s_wait_asynccnt 0x0" ::: "memory");
    __syncthreads();
  }

  if (!active) return;
  int col = (nt << 4) + l16;
  float bv = bias ? bias[col] : 0.f;
#pragma unroll
  for (int mi = 0; mi < 4; ++mi) {
#pragma unroll
    for (int r = 0; r < 8; ++r) {
      int row = mBase + (mi << 4) + (half << 3) + r;
      float x = acc[mi][r] + bv;
      if (act == 1) x = siluf(x);
      if (outF) outF[(size_t)row * Nn + col] = x;
      if (outB) outB[(size_t)row * Nn + col] = tobf(x);
    }
  }
}

// --------------------------- helper kernels --------------------------------

__global__ void scan_kernel(const unsigned char* __restrict__ gm,
                            int* __restrict__ left, int* __restrict__ right) {
  if (blockIdx.x != 0 || threadIdx.x != 0) return;
  int cur = -1;
  for (int i = 0; i < N_; ++i) { if (!gm[i]) cur = i; left[i] = cur; }
  cur = N_;
  for (int i = N_ - 1; i >= 0; --i) { if (!gm[i]) cur = i; right[i] = cur; }
}

__global__ void initmask_kernel(const float* __restrict__ X,
                                const unsigned char* __restrict__ gm,
                                const int* __restrict__ left, const int* __restrict__ right,
                                float* __restrict__ Xcur) {
  int idx = blockIdx.x * blockDim.x + threadIdx.x;
  if (idx >= N_ * C_ * 3) return;
  int n = idx / (C_ * 3), cd = idx % (C_ * 3);
  if (!gm[n]) { Xcur[idx] = X[idx]; return; }
  int lft = left[n], rgt = right[n];
  int li = lft < 0 ? 0 : lft;
  int ri = rgt >= N_ ? N_ - 1 : rgt;
  float denom = (float)max(rgt - lft, 1);
  float t = (float)(n - lft) / denom;
  float xl = X[(size_t)li * 12 + cd], xr = X[(size_t)ri * 12 + cd];
  Xcur[idx] = xl + t * (xr - xl);
}

__global__ void knn_kernel(const float* __restrict__ Xc, const int* __restrict__ seg,
                           const unsigned char* __restrict__ am,
                           int* __restrict__ srcOut, int same) {
  int b = blockIdx.y, i = blockIdx.x, j = threadIdx.x;
  __shared__ float d[L_];
  __shared__ float rv[L_];
  __shared__ int   ri[L_];
  int gi = b * L_ + i, gj = b * L_ + j;
  float best = __builtin_inff();
  for (int c = 0; c < C_; ++c) {
    if (am[gi * C_ + c] && am[gj * C_ + c]) {
      float s = 0.f;
      for (int dd = 0; dd < 3; ++dd) {
        float q = Xc[(size_t)(gi * C_ + c) * 3 + dd] - Xc[(size_t)(gj * C_ + c) * 3 + dd];
        s += q * q;
      }
      best = fminf(best, s);
    }
  }
  bool ss = (seg[gi] == seg[gj]);
  bool cand = (same ? ss : !ss) && (i != j);
  d[j] = cand ? best : __builtin_inff();
  __syncthreads();
  for (int kk = 0; kk < K_; ++kk) {
    rv[j] = d[j]; ri[j] = j;
    __syncthreads();
    for (int s = L_ / 2; s > 0; s >>= 1) {
      if (j < s) {
        if (rv[j + s] < rv[j] || (rv[j + s] == rv[j] && ri[j + s] < ri[j])) {
          rv[j] = rv[j + s]; ri[j] = ri[j + s];
        }
      }
      __syncthreads();
    }
    int jm = ri[0];
    if (j == 0) srcOut[(size_t)gi * K_ + kk] = b * L_ + jm;
    __syncthreads();
    if (j == jm) d[j] = __builtin_inff();
    __syncthreads();
  }
}

__global__ void h_init_kernel(const int* __restrict__ S, const int* __restrict__ pid,
                              const float* __restrict__ aa, const float* __restrict__ pe,
                              const float* __restrict__ re, int r, float* __restrict__ h) {
  int idx = blockIdx.x * blockDim.x + threadIdx.x;
  if (idx >= N_ * HID) return;
  int n = idx >> 7, f = idx & 127;
  h[idx] = aa[(size_t)S[n] * HID + f] + pe[(size_t)pid[n] * HID + f] + re[(size_t)r * HID + f];
}

__global__ void copy_kernel(const float* __restrict__ s, float* __restrict__ dst, int n) {
  int idx = blockIdx.x * blockDim.x + threadIdx.x;
  if (idx < n) dst[idx] = s[idx];
}
__global__ void add_kernel(float* __restrict__ a, const float* __restrict__ b, int n) {
  int idx = blockIdx.x * blockDim.x + threadIdx.x;
  if (idx < n) a[idx] += b[idx];
}

__global__ void edge_feat_kernel(const float* __restrict__ h, const float* __restrict__ Z,
                                 const int* __restrict__ src,
                                 bf16* __restrict__ feat, float* __restrict__ xd) {
  int e = blockIdx.x;                 // edge = n*K + k
  int n = e / K_;
  int s = src[e];
  int t = threadIdx.x;
  __shared__ float sxd[C_ * C_ * 3];
  if (t < C_ * C_ * 3) {
    int dd = t % 3, ab = t / 3, a = ab >> 2, bb = ab & 3;
    float v = Z[(size_t)(n * C_ + a) * 3 + dd] - Z[(size_t)(s * C_ + bb) * 3 + dd];
    sxd[t] = v;
    xd[(size_t)e * 48 + t] = v;
  }
  __syncthreads();
  bf16* fr = feat + (size_t)e * EINP;
  for (int f = t; f < EINP; f += blockDim.x) {
    float v;
    if (f < HID)            v = h[(size_t)n * HID + f];
    else if (f < 2 * HID)   v = h[(size_t)s * HID + (f - HID)];
    else if (f < EIN) {
      int ab = f - 2 * HID;
      float acc = 0.f;
      for (int dd = 0; dd < 3; ++dd) { float q = sxd[ab * 3 + dd]; acc += q * q; }
      v = acc;
    } else v = 0.f;
    fr[f] = tobf(v);
  }
}

__global__ void gate_kernel(float* __restrict__ m, const float* __restrict__ attw,
                            bf16* __restrict__ mg_bf) {
  int e = blockIdx.x, t = threadIdx.x;   // 128 threads
  __shared__ float red[HID];
  float v = m[(size_t)e * HID + t];
  red[t] = v * attw[t];
  __syncthreads();
  for (int s = HID / 2; s > 0; s >>= 1) { if (t < s) red[t] += red[t + s]; __syncthreads(); }
  float g = 1.f / (1.f + __expf(-red[0]));
  float mv = v * g;
  m[(size_t)e * HID + t] = mv;
  mg_bf[(size_t)e * HID + t] = tobf(mv);
}

__global__ void mean_kernel(const float* __restrict__ mg, float* __restrict__ msum, int accF) {
  int idx = blockIdx.x * blockDim.x + threadIdx.x;
  if (idx >= N_ * HID) return;
  int n = idx >> 7, f = idx & 127;
  float s = 0.f;
  for (int k = 0; k < K_; ++k) s += mg[((size_t)n * K_ + k) * HID + f];
  s *= (1.f / K_);
  if (accF) msum[idx] += s; else msum[idx] = s;
}

__global__ void coord_kernel(const float* __restrict__ xd, const float* __restrict__ cw,
                             float* __restrict__ Xupd, int accF) {
  int idx = blockIdx.x * blockDim.x + threadIdx.x;
  if (idx >= N_ * C_ * 3) return;
  int n = idx / 12, ad = idx % 12, a = ad / 3, dd = ad % 3;
  float s = 0.f;
  for (int k = 0; k < K_; ++k) {
    size_t e = (size_t)n * K_ + k;
    for (int bb = 0; bb < C_; ++bb) {
      int ab = a * C_ + bb;
      s += xd[(e * 16 + ab) * 3 + dd] * cw[e * 16 + ab];
    }
  }
  s *= (1.f / (K_ * C_));
  if (accF) Xupd[idx] += s; else Xupd[idx] = s;
}

__global__ void nin_kernel(const float* __restrict__ h, const float* __restrict__ msum,
                           bf16* __restrict__ nin) {
  int idx = blockIdx.x * blockDim.x + threadIdx.x;
  if (idx >= N_ * 2 * HID) return;
  int n = idx >> 8, f = idx & 255;
  float v = (f < HID) ? h[(size_t)n * HID + f] : msum[(size_t)n * HID + (f - HID)];
  nin[idx] = tobf(v);
}

__global__ void where_kernel(const unsigned char* __restrict__ gm,
                             const float* __restrict__ Z, float* __restrict__ Xcur) {
  int idx = blockIdx.x * blockDim.x + threadIdx.x;
  if (idx >= N_ * C_ * 3) return;
  if (gm[idx / 12]) Xcur[idx] = Z[idx];
}

__global__ void loss_partial_kernel(const float* __restrict__ Z, const float* __restrict__ X,
                                    const unsigned char* __restrict__ gm,
                                    float* __restrict__ pa, float* __restrict__ pc) {
  __shared__ float sa[256], sc[256];
  int t = threadIdx.x;
  float a = 0.f, c = 0.f;
  for (int idx = blockIdx.x * 256 + t; idx < N_ * 12; idx += gridDim.x * 256) {
    int n = idx / 12;
    if (gm[n]) {
      float df = Z[idx] - X[idx];
      a += df * df;
      if (idx % 12 == 0) c += 1.f;
    }
  }
  sa[t] = a; sc[t] = c;
  __syncthreads();
  for (int s = 128; s > 0; s >>= 1) { if (t < s) { sa[t] += sa[t + s]; sc[t] += sc[t + s]; } __syncthreads(); }
  if (t == 0) { pa[blockIdx.x] = sa[0]; pc[blockIdx.x] = sc[0]; }
}

__global__ void loss_final_kernel(const float* __restrict__ pa, const float* __restrict__ pc,
                                  float* __restrict__ out) {
  __shared__ float sa[256], sc[256];
  int t = threadIdx.x;
  sa[t] = pa[t]; sc[t] = pc[t];
  __syncthreads();
  for (int s = 128; s > 0; s >>= 1) { if (t < s) { sa[t] += sa[t + s]; sc[t] += sc[t + s]; } __syncthreads(); }
  if (t == 0) out[0] = sa[0] / (sc[0] * (float)C_);
}

// ------------------------------- host side ---------------------------------

static inline void run_gemm(const bf16* A, const bf16* Bp, const float* bias,
                            float* outF, bf16* outB, int M, int Kp, int Nn, int act,
                            hipStream_t s) {
  dim3 grid(M / 64, (Nn / 16 + 7) / 8);
  gemm_bf16_wmma<<<grid, 256, 0, s>>>(A, Bp, bias, outF, outB, M, Kp, Nn, act);
}
static inline void run_pack(const float* W, bf16* out, int Kreal, int Kp, int Nn,
                            hipStream_t s) {
  int total = (Nn / 16) * (Kp / 32) * 512;
  pack_b_kernel<<<(total + 255) / 256, 256, 0, s>>>(W, out, Kreal, Kp, Nn);
}

extern "C" void kernel_launch(void* const* d_in, const int* in_sizes, int n_in,
                              void* d_out, int out_size, void* d_ws, size_t ws_size,
                              hipStream_t stream) {
  (void)in_sizes; (void)n_in; (void)out_size; (void)ws_size;

  const float*         X     = (const float*)d_in[0];
  const int*           S     = (const int*)d_in[1];
  const unsigned char* gmask = (const unsigned char*)d_in[2];
  const int*           pid   = (const int*)d_in[3];
  const int*           segid = (const int*)d_in[4];
  const unsigned char* amask = (const unsigned char*)d_in[6];
  const float*         aa_emb = (const float*)d_in[7];
  const float*         pos_emb = (const float*)d_in[8 + 22 * NLAYERS];
  const float*         rnd_emb = (const float*)d_in[9 + 22 * NLAYERS];
  auto LP = [&](int l, int off) { return (const float*)d_in[8 + 22 * l + off]; };

  // --- workspace carve ---
  char* wp = (char*)d_ws;
  size_t off = 0;
  auto alloc = [&](size_t bytes) -> void* {
    void* p = wp + off;
    off = (off + bytes + 255) & ~(size_t)255;
    return p;
  };
  int*   left  = (int*)alloc(N_ * 4);
  int*   right = (int*)alloc(N_ * 4);
  float* Xcur  = (float*)alloc((size_t)N_ * 12 * 4);
  float* Z     = (float*)alloc((size_t)N_ * 12 * 4);
  float* Xupd  = (float*)alloc((size_t)N_ * 12 * 4);
  int*   srcC  = (int*)alloc((size_t)NK * 4);
  int*   srcI  = (int*)alloc((size_t)NK * 4);
  float* h     = (float*)alloc((size_t)N_ * HID * 4);
  float* msum  = (float*)alloc((size_t)N_ * HID * 4);
  bf16*  featB = (bf16*)alloc((size_t)NK * EINP * 2);   // reused: c1_bf16, nin
  float* xd    = (float*)alloc((size_t)NK * 48 * 4);
  float* mF    = (float*)alloc((size_t)NK * HID * 4);   // m f32 (gated in place); reused: dh
  bf16*  mB    = (bf16*)alloc((size_t)NK * HID * 2);    // m1; reused: mg_bf16, n1
  float* cw    = (float*)alloc((size_t)NK * 16 * 4);
  float* pa    = (float*)alloc(256 * 4);
  float* pc    = (float*)alloc(256 * 4);

  auto packElems = [](int Kp, int Nn) { return (size_t)(Nn / 16) * (Kp / 32) * 512; };
  bf16 *eW0[NLAYERS][2], *eW1[NLAYERS][2], *cW0[NLAYERS][2], *cW1[NLAYERS][2];
  bf16 *nW0[NLAYERS], *nW1[NLAYERS];
  for (int l = 0; l < NLAYERS; ++l) {
    for (int p = 0; p < 2; ++p) {
      eW0[l][p] = (bf16*)alloc(packElems(EINP, HID) * 2);
      eW1[l][p] = (bf16*)alloc(packElems(HID, HID) * 2);
      cW0[l][p] = (bf16*)alloc(packElems(HID, HID) * 2);
      cW1[l][p] = (bf16*)alloc(packElems(HID, 16) * 2);
    }
    nW0[l] = (bf16*)alloc(packElems(2 * HID, HID) * 2);
    nW1[l] = (bf16*)alloc(packElems(HID, HID) * 2);
  }

  // --- pre-swizzle all weights ---
  for (int l = 0; l < NLAYERS; ++l) {
    run_pack(LP(l, 10), eW0[l][0], EIN, EINP, HID, stream);
    run_pack(LP(l, 12), eW1[l][0], HID, HID, HID, stream);
    run_pack(LP(l, 2),  cW0[l][0], HID, HID, HID, stream);
    run_pack(LP(l, 4),  cW1[l][0], HID, HID, 16, stream);
    run_pack(LP(l, 14), eW0[l][1], EIN, EINP, HID, stream);
    run_pack(LP(l, 16), eW1[l][1], HID, HID, HID, stream);
    run_pack(LP(l, 6),  cW0[l][1], HID, HID, HID, stream);
    run_pack(LP(l, 8),  cW1[l][1], HID, HID, 16, stream);
    run_pack(LP(l, 18), nW0[l], 2 * HID, 2 * HID, HID, stream);
    run_pack(LP(l, 20), nW1[l], HID, HID, HID, stream);
  }

  // --- init mask interpolation ---
  scan_kernel<<<1, 1, 0, stream>>>(gmask, left, right);
  initmask_kernel<<<(N_ * 12 + 255) / 256, 256, 0, stream>>>(X, gmask, left, right, Xcur);

  for (int r = 0; r < NITER; ++r) {
    knn_kernel<<<dim3(L_, B_), L_, 0, stream>>>(Xcur, segid, amask, srcC, 1);
    knn_kernel<<<dim3(L_, B_), L_, 0, stream>>>(Xcur, segid, amask, srcI, 0);
    h_init_kernel<<<(N_ * HID + 255) / 256, 256, 0, stream>>>(S, pid, aa_emb, pos_emb,
                                                              rnd_emb, r, h);
    copy_kernel<<<(N_ * 12 + 255) / 256, 256, 0, stream>>>(Xcur, Z, N_ * 12);

    for (int l = 0; l < NLAYERS; ++l) {
      for (int p = 0; p < 2; ++p) {
        const int*   src  = p ? srcI : srcC;
        const float* attw = LP(l, p);                       // att_ctx / att_inter
        const float* eb0 = LP(l, p ? 15 : 11), *eb1 = LP(l, p ? 17 : 13);
        const float* cb0 = LP(l, p ? 7 : 3),   *cb1 = LP(l, p ? 9 : 5);
        edge_feat_kernel<<<NK, 256, 0, stream>>>(h, Z, src, featB, xd);
        run_gemm(featB, eW0[l][p], eb0, nullptr, mB, NK, EINP, HID, 1, stream);
        run_gemm(mB, eW1[l][p], eb1, mF, nullptr, NK, HID, HID, 1, stream);
        gate_kernel<<<NK, HID, 0, stream>>>(mF, attw, mB);
        mean_kernel<<<(N_ * HID + 255) / 256, 256, 0, stream>>>(mF, msum, p);
        run_gemm(mB, cW0[l][p], cb0, nullptr, featB, NK, HID, HID, 1, stream);
        run_gemm(featB, cW1[l][p], cb1, cw, nullptr, NK, HID, 16, 0, stream);
        coord_kernel<<<(N_ * 12 + 255) / 256, 256, 0, stream>>>(xd, cw, Xupd, p);
      }
      // node MLP + residual
      nin_kernel<<<(N_ * 2 * HID + 255) / 256, 256, 0, stream>>>(h, msum, featB);
      run_gemm(featB, nW0[l], LP(l, 19), nullptr, mB, N_, 2 * HID, HID, 1, stream);
      run_gemm(mB, nW1[l], LP(l, 21), mF, nullptr, N_, HID, HID, 0, stream);
      add_kernel<<<(N_ * HID + 255) / 256, 256, 0, stream>>>(h, mF, N_ * HID);
      add_kernel<<<(N_ * 12 + 255) / 256, 256, 0, stream>>>(Z, Xupd, N_ * 12);
    }
    where_kernel<<<(N_ * 12 + 255) / 256, 256, 0, stream>>>(gmask, Z, Xcur);
  }

  loss_partial_kernel<<<256, 256, 0, stream>>>(Z, X, gmask, pa, pc);
  loss_final_kernel<<<1, 256, 0, stream>>>(pa, pc, (float*)d_out);
}
